// AdaptConv2d_42125039239351
// MI455X (gfx1250) — matrix operand order repaired and checked by reading, not compiled, not run on hardware
//
#include <hip/hip_runtime.h>
#include <math.h>

typedef __attribute__((ext_vector_type(16))) __bf16 v16bf;
typedef __attribute__((ext_vector_type(8)))  float  v8f;
typedef __attribute__((ext_vector_type(4)))  int    i4;

typedef __attribute__((address_space(3))) void lds_void;

#define HH     56
#define BHW    3136       // 56*56
#define CIN    256
#define KTOT   2304       // 256*9
#define PDIM   58         // padded spatial
#define PHW    3364       // 58*58
#define NPAD   27541504   // 32*256*3362 (less than 32*256*3364)
#define CG_OH  27
#define CG_OHW 729
#define CG_M   23328      // 32*729
#define ASTR   40         // LDS u16 row stride (80B: 16B-aligned b128 frag reads)

__device__ __forceinline__ unsigned short f2bfu(float f) {
    union { float f; unsigned u; } a; a.f = f;
    return (unsigned short)((a.u + 0x7FFFu + ((a.u >> 16) & 1u)) >> 16);   // RNE
}
__device__ __forceinline__ float sigm(float x) { return 1.0f / (1.0f + __expf(-x)); }

#define WMMA_BF16(A, B, C) \
    __builtin_amdgcn_wmma_f32_16x16x32_bf16(false, (A), false, (B), (short)0, (C), false, false)

// 16B async copy global -> LDS (ASYNCcnt path)
__device__ __forceinline__ void async_ld_b128(void* lds, const void* g) {
#if __has_builtin(__builtin_amdgcn_global_load_async_to_lds_b128)
    __builtin_amdgcn_global_load_async_to_lds_b128((i4*)g, (i4*)lds, 0, 0);
#else
    unsigned off = (unsigned)(unsigned long long)(lds_void*)lds;
    asm volatile("global_load_async_to_lds_b128 %0, %1, off" :: "v"(off), "v"(g) : "memory");
#endif
}
__device__ __forceinline__ void wait_async0() {
#if __has_builtin(__builtin_amdgcn_s_wait_asynccnt)
    __builtin_amdgcn_s_wait_asynccnt(0);
#else
    asm volatile("s_wait_asynccnt 0x0" ::: "memory");
#endif
}

// ---------------- prep kernels ----------------

__global__ void k_zero(float* __restrict__ g, int n) {
    int i = blockIdx.x * 256 + threadIdx.x;
    if (i < n) g[i] = 0.0f;
}

// x (NCHW f32) -> zero-padded bf16 [b][c][58][58]
__global__ void k_cvt_x(const float* __restrict__ x, unsigned short* __restrict__ xp) {
    int idx = blockIdx.x * 256 + threadIdx.x;
    if (idx >= NPAD) return;
    int bc = idx / PHW, rem = idx - bc * PHW;
    int ih = rem / PDIM, iw = rem - ih * PDIM;
    float v = 0.0f;
    if (ih >= 1 && ih <= HH && iw >= 1 && iw <= HH)
        v = x[(size_t)bc * BHW + (ih - 1) * HH + (iw - 1)];
    xp[idx] = f2bfu(v);
}

// conv_w / cg_conv_w (OIHW f32) -> bf16 [o][k], k = (kh*3+kw)*256 + c
__global__ void k_cvt_w(const float* __restrict__ wmain, const float* __restrict__ wcg,
                        unsigned short* __restrict__ wtm, unsigned short* __restrict__ wtc) {
    int idx = blockIdx.x * 256 + threadIdx.x;
    const float* src = wmain; unsigned short* dst = wtm;
    int id = idx;
    if (id >= CIN * KTOT) { id -= CIN * KTOT; src = wcg; dst = wtc; }
    int o = id / KTOT, k = id - o * KTOT;
    int r = k >> 8, c = k & 255;
    int kh = r / 3, kw = r - kh * 3;
    dst[o * KTOT + k] = f2bfu(src[((o * CIN + c) * 3 + kh) * 3 + kw]);
}

// p[b][c] = spatial mean of x
__global__ void k_mean(const float* __restrict__ x, float* __restrict__ p) {
    __shared__ float red[256];
    int bc = blockIdx.x;
    const float* xv = x + (size_t)bc * BHW;
    float s = 0.0f;
    for (int i = threadIdx.x; i < BHW; i += 256) s += xv[i];
    red[threadIdx.x] = s; __syncthreads();
    for (int st = 128; st > 0; st >>= 1) {
        if (threadIdx.x < st) red[threadIdx.x] += red[threadIdx.x + st];
        __syncthreads();
    }
    if (threadIdx.x == 0) p[bc] = red[0] * (1.0f / (float)BHW);
}

// Block tile: 64(M) x 128(N), K-step 32, double-buffered. 8 waves 2(M) x 4(N), wave tile 32x32.
struct Frag { union { v16bf v; i4 d[2]; }; };

// ---------------- channel-gate conv (stride 2, no pad) + fused mean ----------------

__global__ __launch_bounds__(256)
void k_conv_cg(const unsigned short* __restrict__ xp, const unsigned short* __restrict__ wt,
               const float* __restrict__ bias, float* __restrict__ gacc) {
    __shared__ struct { unsigned short A[2][64 * ASTR]; unsigned short B[2][128 * ASTR]; } sm;
    const int tid = threadIdx.x;
    const int m0 = blockIdx.x * 64, n0 = blockIdx.y * 128;
    const int lane = tid & 31, wid = tid >> 5;
    const int wm = (wid >> 2) * 32, wn = (wid & 3) * 32;
    const int half = lane >> 4, lrow = lane & 15;
    v8f a00 = {0,0,0,0,0,0,0,0}, a01 = {0,0,0,0,0,0,0,0};
    v8f a10 = {0,0,0,0,0,0,0,0}, a11 = {0,0,0,0,0,0,0,0};

    const int ml = tid & 63, klb = tid >> 6;
    int m = m0 + ml;
    int mm = (m < CG_M) ? m : (CG_M - 1);                   // address clamp; tail rows unused
    int bb_ = mm / CG_OHW, hw2 = mm - bb_ * CG_OHW;
    int oh = hw2 / CG_OH, ow = hw2 - oh * CG_OH;
    const int pixbase = bb_ * (CIN * PHW) + (oh * 2 + 1) * PDIM + (ow * 2 + 1);
    int abase[8], aldso[8];
    #pragma unroll
    for (int i = 0; i < 8; ++i) {
        abase[i] = pixbase + (klb + 4 * i) * PHW;
        aldso[i] = ml * ASTR + klb + 4 * i;
    }
    const int nlB = tid >> 1, koB = (tid & 1) << 4;
    const unsigned short* wtb = wt + (size_t)(n0 + nlB) * KTOT + koB;

    unsigned short av[8];
    auto loadA = [&](int kt) {
        int r = kt >> 3, c0 = (kt & 7) << 5;
        int khh = r / 3, kww = r - (r / 3) * 3;
        int koff = c0 * PHW + khh * PDIM + kww;
        #pragma unroll
        for (int i = 0; i < 8; ++i) av[i] = xp[abase[i] + koff];
    };
    auto storeA = [&](int buf) {
        #pragma unroll
        for (int i = 0; i < 8; ++i) sm.A[buf][aldso[i]] = av[i];
    };
    auto asyncB = [&](int kt, int buf) {
        const unsigned short* src = wtb + kt * 32;
        async_ld_b128(&sm.B[buf][nlB * ASTR + koB],     src);
        async_ld_b128(&sm.B[buf][nlB * ASTR + koB + 8], src + 8);
    };

    loadA(0); asyncB(0, 0); storeA(0);
    int p = 0;
    for (int kt = 0; kt < 72; ++kt) {
        wait_async0();
        __syncthreads();
        if (kt < 71) { loadA(kt + 1); asyncB(kt + 1, p ^ 1); }
        Frag fa0, fa1, fb0, fb1;
        const unsigned short* Ab = sm.A[p];
        const unsigned short* Bb = sm.B[p];
        const int ar0 = (wm + lrow) * ASTR, ar1 = ar0 + 16 * ASTR, ko2 = half * 16;
        fa0.d[0] = *(const i4*)&Ab[ar0 + ko2]; fa0.d[1] = *(const i4*)&Ab[ar0 + ko2 + 8];
        fa1.d[0] = *(const i4*)&Ab[ar1 + ko2]; fa1.d[1] = *(const i4*)&Ab[ar1 + ko2 + 8];
        const int br0 = (wn + lrow) * ASTR + half * 16, br1 = br0 + 16 * ASTR;
        fb0.d[0] = *(const i4*)&Bb[br0]; fb0.d[1] = *(const i4*)&Bb[br0 + 8];
        fb1.d[0] = *(const i4*)&Bb[br1]; fb1.d[1] = *(const i4*)&Bb[br1 + 8];
        a00 = WMMA_BF16(fa0.v, fb0.v, a00);
        a01 = WMMA_BF16(fa0.v, fb1.v, a01);
        a10 = WMMA_BF16(fa1.v, fb0.v, a10);
        a11 = WMMA_BF16(fa1.v, fb1.v, a11);
        if (kt < 71) storeA(p ^ 1);
        p ^= 1;
    }
    // relu(conv + bias), accumulate spatial mean per (b, n)
    const int nc0 = n0 + wn + lrow, nc1 = nc0 + 16;
    const float bv0 = bias[nc0], bv1 = bias[nc1];
    #pragma unroll
    for (int i = 0; i < 8; ++i) {
        int me = m0 + wm + i + 8 * half;
        if (me < CG_M) {
            int be = me / CG_OHW;
            atomicAdd(&gacc[be * CIN + nc0], fmaxf(a00[i] + bv0, 0.0f));
            atomicAdd(&gacc[be * CIN + nc1], fmaxf(a01[i] + bv1, 0.0f));
        }
        int me2 = me + 16;
        if (me2 < CG_M) {
            int be = me2 / CG_OHW;
            atomicAdd(&gacc[be * CIN + nc0], fmaxf(a10[i] + bv0, 0.0f));
            atomicAdd(&gacc[be * CIN + nc1], fmaxf(a11[i] + bv1, 0.0f));
        }
    }
}

// ---------------- gate finishers ----------------

__global__ void k_cgfc(const float* __restrict__ gacc, const float* __restrict__ fw,
                       const float* __restrict__ fb, float* __restrict__ ch,
                       float* __restrict__ chsum) {
    __shared__ float gb[256];
    __shared__ float red[256];
    int b = blockIdx.x, j = threadIdx.x;
    gb[j] = gacc[b * 256 + j] * (1.0f / (float)CG_OHW);
    __syncthreads();
    float s = fb[j];
    #pragma unroll 4
    for (int c = 0; c < 256; ++c) s += gb[c] * fw[j * 256 + c];
    float cv = (s > 0.0f) ? 1.0f : 0.0f;   // round(sigmoid(relu(s)))
    ch[b * 256 + j] = cv;
    red[j] = cv; __syncthreads();
    for (int st = 128; st > 0; st >>= 1) {
        if (j < st) red[j] += red[j + st];
        __syncthreads();
    }
    if (j == 0) chsum[b] = red[0];
}

__global__ void k_layer(const float* __restrict__ p, const float* __restrict__ cw,
                        const float* __restrict__ cb, const float* __restrict__ wih,
                        const float* __restrict__ bih, const float* __restrict__ bhh,
                        const float* __restrict__ fw, const float* __restrict__ fb,
                        const float* __restrict__ chsum, float* __restrict__ skipf) {
    int b = threadIdx.x;
    if (b >= 32) return;
    float q[10];
    for (int j = 0; j < 10; ++j) {
        float s = cb[j];
        for (int c = 0; c < 256; ++c) s += p[b * 256 + c] * cw[j * 256 + c];
        q[j] = fmaxf(s, 0.0f);
    }
    float hcell[10];
    for (int k = 0; k < 10; ++k) {
        float gi = bih[k]      + bhh[k];
        float gg = bih[20 + k] + bhh[20 + k];
        float go = bih[30 + k] + bhh[30 + k];
        for (int j = 0; j < 10; ++j) {
            gi += q[j] * wih[k * 10 + j];
            gg += q[j] * wih[(20 + k) * 10 + j];
            go += q[j] * wih[(30 + k) * 10 + j];
        }
        float cc = sigm(gi) * tanhf(gg);
        hcell[k] = sigm(go) * tanhf(cc);
    }
    float z = fb[0];
    for (int j = 0; j < 10; ++j) z += hcell[j] * fw[j];
    float layer = (z > 0.0f) ? 1.0f : 0.0f;
    skipf[b] = (layer == 0.0f || chsum[b] == 0.0f) ? 1.0f : 0.0f;
}

// ---------------- main 3x3 conv (stride 1, pad 1) + fused gating select ----------------

__global__ __launch_bounds__(256)
void k_conv_main(const unsigned short* __restrict__ xp, const float* __restrict__ x,
                 const unsigned short* __restrict__ wt, const float* __restrict__ ch,
                 const float* __restrict__ skipf, float* __restrict__ out) {
    __shared__ union {
        struct { unsigned short A[2][64 * ASTR]; unsigned short B[2][128 * ASTR]; } t;
        float o[128 * 66];
    } sm;
    const int tid = threadIdx.x;
    const int m0 = blockIdx.x * 64, n0 = blockIdx.y * 128;
    const int lane = tid & 31, wid = tid >> 5;
    const int wm = (wid >> 2) * 32, wn = (wid & 3) * 32;
    const int half = lane >> 4, lrow = lane & 15;
    v8f a00 = {0,0,0,0,0,0,0,0}, a01 = {0,0,0,0,0,0,0,0};
    v8f a10 = {0,0,0,0,0,0,0,0}, a11 = {0,0,0,0,0,0,0,0};

    const int ml = tid & 63, klb = tid >> 6;
    const int m = m0 + ml;
    const int bI = m / BHW; int hw2 = m - bI * BHW;
    const int h = hw2 / HH, w = hw2 - h * HH;
    const int pixbase = bI * (CIN * PHW) + h * PDIM + w;    // padded coords: +kh,+kw (0..2)
    int abase[8], aldso[8];
    #pragma unroll
    for (int i = 0; i < 8; ++i) {
        abase[i] = pixbase + (klb + 4 * i) * PHW;
        aldso[i] = ml * ASTR + klb + 4 * i;
    }
    const int nlB = tid >> 1, koB = (tid & 1) << 4;
    const unsigned short* wtb = wt + (size_t)(n0 + nlB) * KTOT + koB;

    unsigned short av[8];
    auto loadA = [&](int kt) {
        int r = kt >> 3, c0 = (kt & 7) << 5;
        int khh = r / 3, kww = r - (r / 3) * 3;
        int koff = c0 * PHW + khh * PDIM + kww;
        #pragma unroll
        for (int i = 0; i < 8; ++i) av[i] = xp[abase[i] + koff];
    };
    auto storeA = [&](int buf) {
        #pragma unroll
        for (int i = 0; i < 8; ++i) sm.t.A[buf][aldso[i]] = av[i];
    };
    auto asyncB = [&](int kt, int buf) {
        const unsigned short* src = wtb + kt * 32;
        async_ld_b128(&sm.t.B[buf][nlB * ASTR + koB],     src);
        async_ld_b128(&sm.t.B[buf][nlB * ASTR + koB + 8], src + 8);
    };

    loadA(0); asyncB(0, 0); storeA(0);
    int p = 0;
    for (int kt = 0; kt < 72; ++kt) {
        wait_async0();
        __syncthreads();
        if (kt < 71) { loadA(kt + 1); asyncB(kt + 1, p ^ 1); }
        Frag fa0, fa1, fb0, fb1;
        const unsigned short* Ab = sm.t.A[p];
        const unsigned short* Bb = sm.t.B[p];
        const int ar0 = (wm + lrow) * ASTR, ar1 = ar0 + 16 * ASTR, ko2 = half * 16;
        fa0.d[0] = *(const i4*)&Ab[ar0 + ko2]; fa0.d[1] = *(const i4*)&Ab[ar0 + ko2 + 8];
        fa1.d[0] = *(const i4*)&Ab[ar1 + ko2]; fa1.d[1] = *(const i4*)&Ab[ar1 + ko2 + 8];
        const int br0 = (wn + lrow) * ASTR + half * 16, br1 = br0 + 16 * ASTR;
        fb0.d[0] = *(const i4*)&Bb[br0]; fb0.d[1] = *(const i4*)&Bb[br0 + 8];
        fb1.d[0] = *(const i4*)&Bb[br1]; fb1.d[1] = *(const i4*)&Bb[br1 + 8];
        a00 = WMMA_BF16(fa0.v, fb0.v, a00);
        a01 = WMMA_BF16(fa0.v, fb1.v, a01);
        a10 = WMMA_BF16(fa1.v, fb0.v, a10);
        a11 = WMMA_BF16(fa1.v, fb1.v, a11);
        if (kt < 71) storeA(p ^ 1);
        p ^= 1;
    }
    __syncthreads();   // tiles dead; union reuse as transpose buffer
    #pragma unroll
    for (int i = 0; i < 8; ++i) {
        int row = wm + i + 8 * half;
        sm.o[(wn + lrow) * 66 + row]           = a00[i];
        sm.o[(wn + 16 + lrow) * 66 + row]      = a01[i];
        sm.o[(wn + lrow) * 66 + row + 16]      = a10[i];
        sm.o[(wn + 16 + lrow) * 66 + row + 16] = a11[i];
    }
    __syncthreads();
    #pragma unroll
    for (int it = 0; it < 32; ++it) {
        int e = tid + it * 256;
        int nl = e >> 6, mle = e & 63;
        int me = m0 + mle, n = n0 + nl;
        int be = me / BHW; int hw3 = me - be * BHW;
        int he = hw3 / HH, we = hw3 - he * HH;
        size_t gi = (((size_t)be * CIN + n) * HH + he) * HH + we;
        float xv = x[gi];
        float yv = sm.o[nl * 66 + mle];
        float cg = ch[be * CIN + n];
        float mixed = cg * yv + (1.0f - cg) * xv;
        out[gi] = (skipf[be] != 0.0f) ? xv : mixed;
    }
}

// ---------------- launcher ----------------

extern "C" void kernel_launch(void* const* d_in, const int* in_sizes, int n_in,
                              void* d_out, int out_size, void* d_ws, size_t ws_size,
                              hipStream_t stream) {
    const float* x         = (const float*)d_in[0];
    const float* conv_w    = (const float*)d_in[1];
    const float* cg_conv_w = (const float*)d_in[2];
    const float* cg_conv_b = (const float*)d_in[3];
    const float* cg_fc_w   = (const float*)d_in[4];
    const float* cg_fc_b   = (const float*)d_in[5];
    const float* lg_conv_w = (const float*)d_in[6];
    const float* lg_conv_b = (const float*)d_in[7];
    const float* w_ih      = (const float*)d_in[8];
    /* d_in[9] = w_hh unused (h0 == 0) */
    const float* b_ih      = (const float*)d_in[10];
    const float* b_hh      = (const float*)d_in[11];
    const float* lg_fc_w   = (const float*)d_in[12];
    const float* lg_fc_b   = (const float*)d_in[13];
    float* out = (float*)d_out;

    char* ws = (char*)d_ws;
    unsigned short* xpad = (unsigned short*)(ws + 0);          // 55,083,008 B
    unsigned short* wtm  = (unsigned short*)(ws + 55083008);   // 1,179,648 B
    unsigned short* wtc  = (unsigned short*)(ws + 56262656);   // 1,179,648 B
    float* p     = (float*)(ws + 57442304);                    // 8192 f32
    float* gacc  = (float*)(ws + 57475072);                    // 8192 f32
    float* chm   = (float*)(ws + 57507840);                    // 8192 f32
    float* chsum = (float*)(ws + 57540608);                    // 32 f32
    float* skipf = (float*)(ws + 57540736);                    // 32 f32

    k_zero  <<<32, 256, 0, stream>>>(gacc, 8192);
    k_cvt_x <<<(NPAD + 255) / 256, 256, 0, stream>>>(x, xpad);
    k_cvt_w <<<4608, 256, 0, stream>>>(conv_w, cg_conv_w, wtm, wtc);
    k_mean  <<<8192, 256, 0, stream>>>(x, p);
    k_conv_cg<<<dim3(365, 2), 256, 0, stream>>>(xpad, wtc, cg_conv_b, gacc);
    k_cgfc  <<<32, 256, 0, stream>>>(gacc, cg_fc_w, cg_fc_b, chm, chsum);
    k_layer <<<1, 32, 0, stream>>>(p, lg_conv_w, lg_conv_b, w_ih, b_ih, b_hh,
                                   lg_fc_w, lg_fc_b, chsum, skipf);
    k_conv_main<<<dim3(1568, 2), 256, 0, stream>>>(xpad, x, wtm, chm, skipf, out);
}
